// NNModel_35356170780676
// MI455X (gfx1250) — matrix operand order repaired and checked by reading, not compiled, not run on hardware
//
#include <hip/hip_runtime.h>
#include <hip/hip_bf16.h>

// ---------------------------------------------------------------------------
// 2-layer LSTM LM on MI455X (gfx1250).
//  - bf16 WMMA (v_wmma_f32_16x16x32_bf16), f32 accumulate
//  - weights pre-packed into exact B-fragment order (contiguous 32B/lane)
//  - activation A panels staged into LDS via the Tensor Data Mover
//    (tensor_load_to_lds + s_wait_tensorcnt), A fragments via ds_load_b128
//  - 4 M-tiles per wave: each B fragment feeds 4 back-to-back WMMAs
// d_in: raw_input(i32), h1,h2,c1,c2, emb, wi1, wh1, wi2, wh2, dec_w, dec_b
// d_out: decoded[100*64*10000] ++ h1f[64*512] ++ h2f ++ c1f ++ c2f (f32)
// ---------------------------------------------------------------------------

typedef __attribute__((ext_vector_type(16))) __bf16 v16bf;
typedef __attribute__((ext_vector_type(8)))  __bf16 v8bf;
typedef __attribute__((ext_vector_type(8)))  float  v8f;
typedef __attribute__((ext_vector_type(4)))  unsigned int v4u;
typedef __attribute__((ext_vector_type(8)))  int    v8i;
typedef __attribute__((ext_vector_type(4)))  int    v4i;

#define HID   512
#define BATCH 64
#define TWIN  100
#define VOC   10000
#define KT_H  16          // 512/32 K-steps
#define NT_H  32          // 512/16 N-tiles
#define NT_V  625         // 10000/16 N-tiles
#define SEQ_ROWS (TWIN*BATCH)   // 6400

#if __has_builtin(__builtin_amdgcn_tensor_load_to_lds)
#define HAVE_TDM 1
#else
#define HAVE_TDM 0
#endif

// ----- fast transcendental helpers (v_exp_f32 / v_rcp_f32) -----------------
__device__ __forceinline__ float fsigmoid(float x) {
    return __builtin_amdgcn_rcpf(1.0f + __expf(-x));
}
__device__ __forceinline__ float ftanh(float x) {
    float e = __expf(-2.0f * x);
    return (1.0f - e) * __builtin_amdgcn_rcpf(1.0f + e);
}

// ----- A-fragment loader: 16x32 bf16 from a row-major panel ----------------
// lane l<16  : row M=l,    K = k0+0..7 (e0..7), k0+16..23 (e8..15)
// lane l>=16 : row M=l-16, K = k0+8..15,         k0+24..31
__device__ __forceinline__ v16bf load_a_frag(const __bf16* p) {
    v8bf lo = *(const v8bf*)(p);
    v8bf hi = *(const v8bf*)(p + 16);
    return __builtin_shufflevector(lo, hi, 0,1,2,3,4,5,6,7,8,9,10,11,12,13,14,15);
}

// ---------------------------------------------------------------------------
// TDM: DMA a 2D bf16 tile [rows x row_elems] (row stride row_stride_elems)
// from global memory into LDS at byte offset lds_off.  D# per ISA 8.3-8.5.
// ---------------------------------------------------------------------------
__device__ __forceinline__ void tdm_load_2d_bf16(unsigned int lds_off, const void* gptr,
                                                 unsigned int rows, unsigned int row_elems,
                                                 unsigned int row_stride_elems) {
#if HAVE_TDM
    unsigned long long ga = (unsigned long long)(uintptr_t)gptr;
    v4u g0;
    g0.x = 1u;                                                    // count=1, user mode
    g0.y = lds_off;                                               // lds_addr (bytes)
    g0.z = (unsigned int)ga;                                      // global_addr[31:0]
    g0.w = (unsigned int)((ga >> 32) & 0x01FFFFFFu) | 0x80000000u; // addr[56:32] | type=2
    v8i g1;
    g1[0] = (int)(1u << 16);                                      // data_size=1 (2 bytes)
    g1[1] = (int)((row_elems & 0xFFFFu) << 16);                   // tensor_dim0[15:0]
    g1[2] = (int)(((row_elems >> 16) & 0xFFFFu) |
                  ((rows & 0xFFFFu) << 16));                      // dim0 hi | tensor_dim1 lo
    g1[3] = (int)(((rows >> 16) & 0xFFFFu) |
                  ((row_elems & 0xFFFFu) << 16));                 // dim1 hi | tile_dim0
    g1[4] = (int)(rows & 0xFFFFu);                                // tile_dim1 (tile_dim2=0)
    g1[5] = (int)row_stride_elems;                                // tensor_dim0_stride lo32
    g1[6] = 0;
    g1[7] = 0;
    v4i z4 = {0, 0, 0, 0};
#if __clang_major__ >= 23
    v8i z8 = {0, 0, 0, 0, 0, 0, 0, 0};
    __builtin_amdgcn_tensor_load_to_lds(g0, g1, z4, z4, z8, 0);
#else
    __builtin_amdgcn_tensor_load_to_lds(g0, g1, z4, z4, 0);
#endif
#endif
}

// Stage a [rows x 512] bf16 panel into LDS (TDM by wave 0, else cooperative).
__device__ __forceinline__ void stage_panel(__bf16* smem, const __bf16* gsrc, int rows) {
#if HAVE_TDM
    if (threadIdx.x == 0) {
        tdm_load_2d_bf16((unsigned int)(uintptr_t)smem, gsrc, (unsigned int)rows,
                         (unsigned int)HID, (unsigned int)HID);
        __builtin_amdgcn_s_wait_tensorcnt(0);
    }
#else
    for (int i = threadIdx.x; i < rows * HID / 8; i += blockDim.x)
        ((v8bf*)smem)[i] = ((const v8bf*)gsrc)[i];
#endif
    __syncthreads();
}

// ---------------------------------------------------------------------------
// Pack a [K=512][N=512] f32 weight (one gate) into B-fragment order, bf16.
// B-frag layout: lane<16 holds K k0+0..7, k0+16..23 for col N=lane; lane+16 -> K k0+8..15, k0+24..31.
// ---------------------------------------------------------------------------
__global__ void pack_kn_kernel(const float* __restrict__ w, __bf16* __restrict__ p) {
    int idx  = blockIdx.x * blockDim.x + threadIdx.x;    // < 512*512
    int e    = idx & 15;
    int lane = (idx >> 4) & 31;
    int kt   = (idx >> 9) & (KT_H - 1);
    int nt   = idx >> 13;
    int K = kt * 32 + (e & 7) + ((lane >> 4) << 3) + ((e >> 3) << 4);
    int N = nt * 16 + (lane & 15);
    p[idx] = (__bf16)w[K * HID + N];
}

// dec_w is [N=10000][K=512] row-major: B[k][n] = dec_w[n][k]
__global__ void pack_nk_kernel(const float* __restrict__ w, __bf16* __restrict__ p, int total) {
    int idx = blockIdx.x * blockDim.x + threadIdx.x;
    if (idx >= total) return;
    int e    = idx & 15;
    int lane = (idx >> 4) & 31;
    int kt   = (idx >> 9) & (KT_H - 1);
    int nt   = idx >> 13;                                // 0..624
    int K = kt * 32 + (e & 7) + ((lane >> 4) << 3) + ((e >> 3) << 4);
    int N = nt * 16 + (lane & 15);
    p[idx] = (__bf16)w[N * HID + K];
}

// ----- embedding gather, f32 -> bf16 row-major [6400][512] ------------------
__global__ void embed_kernel(const int* __restrict__ tok, const float* __restrict__ emb,
                             __bf16* __restrict__ x) {
    int row = blockIdx.x;            // t*64 + b
    int i   = threadIdx.x;           // 0..511
    int t   = tok[row];
    x[row * HID + i] = (__bf16)emb[t * HID + i];
}

// ----- initial state: h f32 -> bf16 state buf + f32 mirror; c -> f32 region -
__global__ void init_state_kernel(const float* __restrict__ h0, const float* __restrict__ c0,
                                  __bf16* __restrict__ hb, float* __restrict__ hf,
                                  float* __restrict__ cf) {
    int i = blockIdx.x * blockDim.x + threadIdx.x;       // < 64*512
    float h = h0[i];
    hb[i] = (__bf16)h;
    hf[i] = h;
    cf[i] = c0[i];
}

// ---------------------------------------------------------------------------
// One fused LSTM layer step.  Grid: 8 blocks x 128 threads (4 waves).
// Each wave owns one N-tile (16 hidden cols) and ALL 4 M-tiles (full batch):
// every B fragment (1KB weights) feeds 4 back-to-back WMMAs per gate.
// The 64x512 x-panel is TDM-staged into LDS once per block; h reads are
// global (64KB, L2-hot).  Total per step: weights read exactly once.
// ---------------------------------------------------------------------------
__global__ __launch_bounds__(128, 1)
void lstm_step_kernel(const __bf16* __restrict__ x,     // [64][512]
                      const __bf16* __restrict__ hin,   // [64][512]
                      const __bf16* __restrict__ wip,   // packed [4][512*512]
                      const __bf16* __restrict__ whp,   // packed [4][512*512]
                      float*  __restrict__ c,           // [64][512] in-place
                      __bf16* __restrict__ hout,        // [64][512]
                      float*  __restrict__ houtf,       // [64][512] f32 mirror
                      __bf16* __restrict__ seq)         // nullable
{
    __shared__ __align__(64) __bf16 smx[BATCH * HID];   // 64 KB

    int widx = threadIdx.x >> 5;
    int lane = threadIdx.x & 31;
    int nt   = blockIdx.x * 4 + widx;        // 0..31

    stage_panel(smx, x, BATCH);

    const __bf16* xbase = smx + (lane & 15) * HID + ((lane >> 4) << 3);
    const __bf16* hbase = hin + (lane & 15) * HID + ((lane >> 4) << 3);

    v8f acc[4][4] = {};                      // [gate][mtile]

    // gates += x * Wi
    for (int kt = 0; kt < KT_H; ++kt) {
        v16bf b[4];
        #pragma unroll
        for (int g = 0; g < 4; ++g)
            b[g] = *(const v16bf*)(wip + ((size_t)(((g * NT_H + nt) * KT_H + kt) * 32 + lane)) * 16);
        #pragma unroll
        for (int m = 0; m < 4; ++m) {
            v16bf a = load_a_frag(xbase + m * 16 * HID + kt * 32);   // ds_load_b128 x2
            #pragma unroll
            for (int g = 0; g < 4; ++g)
                acc[g][m] = __builtin_amdgcn_wmma_f32_16x16x32_bf16(false, a, false, b[g],
                                                                    (short)0, acc[g][m],
                                                                    false, false);
        }
    }
    // gates += h * Wh
    for (int kt = 0; kt < KT_H; ++kt) {
        v16bf b[4];
        #pragma unroll
        for (int g = 0; g < 4; ++g)
            b[g] = *(const v16bf*)(whp + ((size_t)(((g * NT_H + nt) * KT_H + kt) * 32 + lane)) * 16);
        #pragma unroll
        for (int m = 0; m < 4; ++m) {
            v16bf a = load_a_frag(hbase + m * 16 * HID + kt * 32);
            #pragma unroll
            for (int g = 0; g < 4; ++g)
                acc[g][m] = __builtin_amdgcn_wmma_f32_16x16x32_bf16(false, a, false, b[g],
                                                                    (short)0, acc[g][m],
                                                                    false, false);
        }
    }

    // cell update; C/D layout: VGPR r -> M = r + 8*(lane>=16), N = lane&15
    int col = nt * 16 + (lane & 15);
    #pragma unroll
    for (int m = 0; m < 4; ++m) {
        #pragma unroll
        for (int r = 0; r < 8; ++r) {
            int row = m * 16 + r + ((lane >> 4) << 3);
            int idx = row * HID + col;
            float gi = fsigmoid(acc[0][m][r]);
            float gf = fsigmoid(acc[1][m][r]);
            float gg = ftanh(acc[2][m][r]);
            float go = fsigmoid(acc[3][m][r]);
            float cn = gf * c[idx] + gi * gg;
            c[idx] = cn;
            float h = go * ftanh(cn);
            hout[idx]  = (__bf16)h;
            houtf[idx] = h;
            if (seq) seq[idx] = (__bf16)h;
        }
    }
}

// ---------------------------------------------------------------------------
// Decoder: [6400 x 512] x [512 x 10000] + bias, f32 out.
// Grid: (157, 100) x 128 threads.  Each wave: 1 N-tile x 4 M-tiles, so each
// dec_w B fragment feeds 4 WMMAs; 64-row A panel TDM-staged per block.
// ---------------------------------------------------------------------------
__global__ __launch_bounds__(128, 1)
void decoder_kernel(const __bf16* __restrict__ xseq,   // [6400][512]
                    const __bf16* __restrict__ wp,     // packed dec_w
                    const float*  __restrict__ bias,   // [10000]
                    float* __restrict__ out)           // [6400][10000]
{
    __shared__ __align__(64) __bf16 sma[64 * HID];     // 64 KB

    int widx = threadIdx.x >> 5;
    int lane = threadIdx.x & 31;
    int nt   = blockIdx.x * 4 + widx;
    int row0 = blockIdx.y * 64;               // 0..99 groups of 64 rows

    stage_panel(sma, xseq + (size_t)row0 * HID, 64);
    if (nt >= NT_V) return;                   // after barrier; wave-uniform

    const __bf16* abase = sma + (lane & 15) * HID + ((lane >> 4) << 3);

    v8f acc[4] = {};
    for (int kt = 0; kt < KT_H; ++kt) {
        v16bf b = *(const v16bf*)(wp + ((size_t)((nt * KT_H + kt) * 32 + lane)) * 16);
        #pragma unroll
        for (int m = 0; m < 4; ++m) {
            v16bf a = load_a_frag(abase + m * 16 * HID + kt * 32);
            acc[m] = __builtin_amdgcn_wmma_f32_16x16x32_bf16(false, a, false, b,
                                                             (short)0, acc[m], false, false);
        }
    }

    int col = nt * 16 + (lane & 15);
    float bv = bias[col];
    #pragma unroll
    for (int m = 0; m < 4; ++m) {
        #pragma unroll
        for (int r = 0; r < 8; ++r) {
            int row = row0 + m * 16 + r + ((lane >> 4) << 3);
            out[(size_t)row * VOC + col] = acc[m][r] + bv;
        }
    }
}

// ---------------------------------------------------------------------------
extern "C" void kernel_launch(void* const* d_in, const int* in_sizes, int n_in,
                              void* d_out, int out_size, void* d_ws, size_t ws_size,
                              hipStream_t stream) {
    (void)in_sizes; (void)n_in; (void)out_size; (void)ws_size;

    const int*   raw   = (const int*)  d_in[0];
    const float* h1_0  = (const float*)d_in[1];
    const float* h2_0  = (const float*)d_in[2];
    const float* c1_0  = (const float*)d_in[3];
    const float* c2_0  = (const float*)d_in[4];
    const float* emb   = (const float*)d_in[5];
    const float* wi1   = (const float*)d_in[6];
    const float* wh1   = (const float*)d_in[7];
    const float* wi2   = (const float*)d_in[8];
    const float* wh2   = (const float*)d_in[9];
    const float* dec_w = (const float*)d_in[10];
    const float* dec_b = (const float*)d_in[11];

    float* out = (float*)d_out;
    float* h1f = out + (size_t)SEQ_ROWS * VOC;   // 64,000,000
    float* h2f = h1f + BATCH * HID;
    float* c1f = h2f + BATCH * HID;
    float* c2f = c1f + BATCH * HID;

    // ---- workspace carve-up (all offsets 32B-aligned) ----
    char* ws = (char*)d_ws;
    __bf16* xemb = (__bf16*)ws;  ws += (size_t)SEQ_ROWS * HID * 2;   // 6.55 MB
    __bf16* oseq = (__bf16*)ws;  ws += (size_t)SEQ_ROWS * HID * 2;   // 6.55 MB
    __bf16* h1b[2]; __bf16* h2b[2];
    h1b[0] = (__bf16*)ws; ws += BATCH * HID * 2;
    h1b[1] = (__bf16*)ws; ws += BATCH * HID * 2;
    h2b[0] = (__bf16*)ws; ws += BATCH * HID * 2;
    h2b[1] = (__bf16*)ws; ws += BATCH * HID * 2;
    __bf16* wi1p = (__bf16*)ws; ws += (size_t)4 * HID * HID * 2;
    __bf16* wh1p = (__bf16*)ws; ws += (size_t)4 * HID * HID * 2;
    __bf16* wi2p = (__bf16*)ws; ws += (size_t)4 * HID * HID * 2;
    __bf16* wh2p = (__bf16*)ws; ws += (size_t)4 * HID * HID * 2;
    __bf16* decp = (__bf16*)ws; ws += (size_t)VOC * HID * 2;         // 10.24 MB

    // ---- 1) pack weights to bf16 B-fragment order ----
    const float* wsrc[4] = {wi1, wh1, wi2, wh2};
    __bf16*      wdst[4] = {wi1p, wh1p, wi2p, wh2p};
    for (int m = 0; m < 4; ++m)
        for (int g = 0; g < 4; ++g)
            pack_kn_kernel<<<(HID * HID) / 256, 256, 0, stream>>>(
                wsrc[m] + (size_t)g * HID * HID, wdst[m] + (size_t)g * HID * HID);
    pack_nk_kernel<<<(VOC * HID) / 256, 256, 0, stream>>>(dec_w, decp, VOC * HID);

    // ---- 2) embedding gather -> bf16 ----
    embed_kernel<<<SEQ_ROWS, HID, 0, stream>>>(raw, emb, xemb);

    // ---- 3) initial states (c lives in-place in d_out final-state region) ----
    init_state_kernel<<<(BATCH * HID) / 256, 256, 0, stream>>>(h1_0, c1_0, h1b[0], h1f, c1f);
    init_state_kernel<<<(BATCH * HID) / 256, 256, 0, stream>>>(h2_0, c2_0, h2b[0], h2f, c2f);

    // ---- 4) 100 sequential steps, 2 fused layer kernels each ----
    for (int t = 0; t < TWIN; ++t) {
        int cur = t & 1, nxt = cur ^ 1;
        lstm_step_kernel<<<8, 128, 0, stream>>>(
            xemb + (size_t)t * BATCH * HID, h1b[cur], wi1p, wh1p,
            c1f, h1b[nxt], h1f, (__bf16*)nullptr);
        lstm_step_kernel<<<8, 128, 0, stream>>>(
            h1b[nxt], h2b[cur], wi2p, wh2p,
            c2f, h2b[nxt], h2f, oseq + (size_t)t * BATCH * HID);
    }

    // ---- 5) decoder GEMM + bias ----
    decoder_kernel<<<dim3((NT_V + 3) / 4, SEQ_ROWS / 64), 128, 0, stream>>>(
        oseq, decp, dec_b, out);
}
